// GlobalSelfAttentionLayer_67740224193147
// MI455X (gfx1250) — matrix operand order repaired and checked by reading, not compiled, not run on hardware
//
#include <hip/hip_runtime.h>
#include <hip/hip_bf16.h>
#include <math.h>

#define IN_F  768
#define OUT_F 256
#define NTOK  8192
#define ALPHA 0.2f

typedef __attribute__((ext_vector_type(16))) __bf16 v16bf;
typedef __attribute__((ext_vector_type(8)))  float  v8f;

__device__ __forceinline__ unsigned short f32_to_bf16_rne(float f) {
    union { float f; unsigned u; } v; v.f = f;
    unsigned u = v.u;
    unsigned r = u + 0x7FFFu + ((u >> 16) & 1u);   // round-to-nearest-even
    return (unsigned short)(r >> 16);
}

// ---------------------------------------------------------------- convert X
__global__ void __launch_bounds__(256) gsal_cvt_x(
    const float* __restrict__ X, unsigned short* __restrict__ Xb, int n) {
    int i = blockIdx.x * 256 + threadIdx.x;
    if (i < n) Xb[i] = f32_to_bf16_rne(X[i]);
}

// ------------------------------------------- convert W -> bf16, transposed
// Wt[n][k] = bf16(W[k][n]);  W is [IN_F][OUT_F] row-major.
__global__ void __launch_bounds__(256) gsal_cvt_wt(
    const float* __restrict__ W, unsigned short* __restrict__ Wtb) {
    int i = blockIdx.x * 256 + threadIdx.x;        // i over OUT_F*IN_F
    if (i < OUT_F * IN_F) {
        int n = i / IN_F;
        int k = i - n * IN_F;
        Wtb[i] = f32_to_bf16_rne(W[(size_t)k * OUT_F + n]);
    }
}

// ------------------------------------------------------------ WMMA GEMM
// Wh[NTOK][OUT_F] = X[NTOK][IN_F] * W[IN_F][OUT_F], bf16 inputs, f32 accum.
// Block = 128 threads = 4 waves; wave w owns the 16x64 tile (tileM, cols w*64..).
// Software-pipelined K loop: fragments for step k+1 are loaded before the
// 4 WMMAs of step k execute, so loads overlap XDL work within the wave.
__global__ void __launch_bounds__(128) gsal_gemm_wmma(
    const unsigned short* __restrict__ Xb,    // [NTOK][IN_F] bf16
    const unsigned short* __restrict__ Wtb,   // [OUT_F][IN_F] bf16 (transposed W)
    float* __restrict__ Wh) {
    const int wave = threadIdx.x >> 5;
    const int lane = threadIdx.x & 31;
    const int m    = lane & 15;     // A row / B col / D col within 16-tile
    const int h    = lane >> 4;     // lane half selector
    const int tileM = blockIdx.x;   // 0..511
    const int n0    = wave * 64;

    union Frag { v16bf v; uint4 q[2]; };

    v8f acc0 = {}, acc1 = {}, acc2 = {}, acc3 = {};

    const unsigned short* Arow = Xb  + (size_t)(tileM * 16 + m) * IN_F;
    const unsigned short* B0   = Wtb + (size_t)(n0 +  0 + m) * IN_F;
    const unsigned short* B1   = Wtb + (size_t)(n0 + 16 + m) * IN_F;
    const unsigned short* B2   = Wtb + (size_t)(n0 + 32 + m) * IN_F;
    const unsigned short* B3   = Wtb + (size_t)(n0 + 48 + m) * IN_F;

    // ---- prologue: load fragments for k0 = 0
    Frag a, b0, b1, b2, b3;
    a.q[0]  = *(const uint4*)(Arow + h * 16);
    a.q[1]  = *(const uint4*)(Arow + h * 16 + 8);
    {
        const int bk = h * 16;
        b0.q[0] = *(const uint4*)(B0 + bk);  b0.q[1] = *(const uint4*)(B0 + bk + 8);
        b1.q[0] = *(const uint4*)(B1 + bk);  b1.q[1] = *(const uint4*)(B1 + bk + 8);
        b2.q[0] = *(const uint4*)(B2 + bk);  b2.q[1] = *(const uint4*)(B2 + bk + 8);
        b3.q[0] = *(const uint4*)(B3 + bk);  b3.q[1] = *(const uint4*)(B3 + bk + 8);
    }

    #pragma unroll 4
    for (int k0 = 0; k0 < IN_F; k0 += 32) {
        // ---- prefetch next K-step (clamped: last step re-loads itself,
        //      stays in-bounds and keeps the body branch-free)
        int kn = k0 + 32;
        if (kn > IN_F - 32) kn = IN_F - 32;
        Frag an, b0n, b1n, b2n, b3n;
        an.q[0]  = *(const uint4*)(Arow + kn + h * 16);
        an.q[1]  = *(const uint4*)(Arow + kn + h * 16 + 8);
        const int bkn = kn + h * 16;
        b0n.q[0] = *(const uint4*)(B0 + bkn);  b0n.q[1] = *(const uint4*)(B0 + bkn + 8);
        b1n.q[0] = *(const uint4*)(B1 + bkn);  b1n.q[1] = *(const uint4*)(B1 + bkn + 8);
        b2n.q[0] = *(const uint4*)(B2 + bkn);  b2n.q[1] = *(const uint4*)(B2 + bkn + 8);
        b3n.q[0] = *(const uint4*)(B3 + bkn);  b3n.q[1] = *(const uint4*)(B3 + bkn + 8);

        // ---- consume current fragments on the matrix pipe
        acc0 = __builtin_amdgcn_wmma_f32_16x16x32_bf16(false, a.v, false, b0.v, (short)0, acc0, false, false);
        acc1 = __builtin_amdgcn_wmma_f32_16x16x32_bf16(false, a.v, false, b1.v, (short)0, acc1, false, false);
        acc2 = __builtin_amdgcn_wmma_f32_16x16x32_bf16(false, a.v, false, b2.v, (short)0, acc2, false, false);
        acc3 = __builtin_amdgcn_wmma_f32_16x16x32_bf16(false, a.v, false, b3.v, (short)0, acc3, false, false);

        a = an; b0 = b0n; b1 = b1n; b2 = b2n; b3 = b3n;
    }

    // D layout: lane (0-15,16-31) -> col m, rows r+8h for VGPR r
    float* out = Wh + (size_t)(tileM * 16 + 8 * h) * OUT_F + n0 + m;
    #pragma unroll
    for (int r = 0; r < 8; ++r) {
        size_t ro = (size_t)r * OUT_F;
        out[ro +  0] = acc0[r];
        out[ro + 16] = acc1[r];
        out[ro + 32] = acc2[r];
        out[ro + 48] = acc3[r];
    }
}

// ----------------------------------------- s1[i] = Wh[i,:].a1, s2[i] = Wh[i,:].a2
__global__ void __launch_bounds__(256) gsal_rowdot(
    const float* __restrict__ Wh, const float* __restrict__ a,
    float* __restrict__ s1, float* __restrict__ s2) {
    const int wave = threadIdx.x >> 5;
    const int lane = threadIdx.x & 31;
    const int row  = blockIdx.x * 8 + wave;
    const float* wr = Wh + (size_t)row * OUT_F;
    float acc1 = 0.f, acc2 = 0.f;
    #pragma unroll
    for (int c = lane; c < OUT_F; c += 32) {
        float v = wr[c];
        acc1 = fmaf(v, a[c],         acc1);
        acc2 = fmaf(v, a[OUT_F + c], acc2);
    }
    #pragma unroll
    for (int off = 16; off > 0; off >>= 1) {
        acc1 += __shfl_xor(acc1, off, 32);
        acc2 += __shfl_xor(acc2, off, 32);
    }
    if (lane == 0) { s1[row] = acc1; s2[row] = acc2; }
}

// --------------------------- r[i] = sum_j leakyrelu(s1[i] + s2[j])  (N^2 pass)
__global__ void __launch_bounds__(256) gsal_rowsums(
    const float* __restrict__ s1, const float* __restrict__ s2,
    float* __restrict__ rsum) {
    __shared__ float sh[NTOK];                     // 32 KB of the 320 KB LDS
    for (int j = threadIdx.x; j < NTOK; j += 256) sh[j] = s2[j];
    __syncthreads();
    const int i = blockIdx.x * 256 + threadIdx.x;
    const float si = s1[i];
    float acc = 0.f;
    #pragma unroll 8
    for (int j = 0; j < NTOK; ++j) {
        float t = si + sh[j];                      // LDS broadcast read
        acc += (t > 0.f) ? t : (ALPHA * t);
    }
    rsum[i] = acc;
}

// ------------------------------------------- att = softmax(rsum) (single block)
__global__ void __launch_bounds__(1024) gsal_softmax(
    const float* __restrict__ rsum, float* __restrict__ att) {
    __shared__ float red[1024];
    const int t = threadIdx.x;
    float m = -INFINITY;
    for (int i = t; i < NTOK; i += 1024) m = fmaxf(m, rsum[i]);
    red[t] = m; __syncthreads();
    for (int s = 512; s > 0; s >>= 1) {
        if (t < s) red[t] = fmaxf(red[t], red[t + s]);
        __syncthreads();
    }
    m = red[0]; __syncthreads();
    float z = 0.f;
    for (int i = t; i < NTOK; i += 1024) z += expf(rsum[i] - m);
    red[t] = z; __syncthreads();
    for (int s = 512; s > 0; s >>= 1) {
        if (t < s) red[t] += red[t + s];
        __syncthreads();
    }
    const float invZ = 1.0f / red[0];
    for (int i = t; i < NTOK; i += 1024) att[i] = expf(rsum[i] - m) * invZ;
}

// ------------------------- partial[b][f] = sum_{i in block b} att[i]*Wh[i][f]
__global__ void __launch_bounds__(256) gsal_attn_partial(
    const float* __restrict__ Wh, const float* __restrict__ att,
    float* __restrict__ partial) {
    const int b  = blockIdx.x;          // 32 blocks, 256 rows each
    const int t  = threadIdx.x;         // column
    const int i0 = b * 256;
    float acc = 0.f;
    #pragma unroll 4
    for (int i = 0; i < 256; ++i) {
        float w = att[i0 + i];          // uniform -> scalar broadcast
        acc = fmaf(w, Wh[(size_t)(i0 + i) * OUT_F + t], acc);
    }
    partial[(size_t)b * OUT_F + t] = acc;
}

// --------------------------- out[f] = elu(sum_b partial[b][f]) (fixed order)
__global__ void __launch_bounds__(256) gsal_finalize(
    const float* __restrict__ partial, float* __restrict__ out) {
    const int t = threadIdx.x;
    float acc = 0.f;
    #pragma unroll
    for (int p = 0; p < 32; ++p) acc += partial[(size_t)p * OUT_F + t];
    out[t] = (acc > 0.f) ? acc : expm1f(acc);     // elu, alpha=1
}

extern "C" void kernel_launch(void* const* d_in, const int* in_sizes, int n_in,
                              void* d_out, int out_size, void* d_ws, size_t ws_size,
                              hipStream_t stream) {
    (void)in_sizes; (void)n_in; (void)out_size; (void)ws_size;
    const float* X = (const float*)d_in[0];   // [NTOK*IN_F]
    const float* W = (const float*)d_in[1];   // [IN_F*OUT_F]
    const float* a = (const float*)d_in[2];   // [2*OUT_F]
    float* out = (float*)d_out;               // [OUT_F]

    // Workspace layout (bytes, all 16B-aligned)
    char* ws = (char*)d_ws;
    unsigned short* Xb  = (unsigned short*)(ws);                         // 12,582,912 B
    unsigned short* Wtb = (unsigned short*)(ws + 12582912);              //    393,216 B
    float* Wh      = (float*)(ws + 12976128);                            //  8,388,608 B
    float* s1      = (float*)(ws + 21364736);                            //     32,768 B
    float* s2      = (float*)(ws + 21397504);                            //     32,768 B
    float* rsum    = (float*)(ws + 21430272);                            //     32,768 B
    float* att     = (float*)(ws + 21463040);                            //     32,768 B
    float* partial = (float*)(ws + 21495808);                            //     32,768 B

    // 1. precision conversion (bf16 inputs, fp32 accumulate in WMMA)
    gsal_cvt_x <<<(NTOK * IN_F + 255) / 256, 256, 0, stream>>>(X, Xb, NTOK * IN_F);
    gsal_cvt_wt<<<(OUT_F * IN_F + 255) / 256, 256, 0, stream>>>(W, Wtb);

    // 2. Wh = X @ W on the matrix pipe (pipelined WMMA)
    gsal_gemm_wmma<<<NTOK / 16, 128, 0, stream>>>(Xb, Wtb, Wh);

    // 3. s1/s2 row dots (wave-per-row)
    gsal_rowdot<<<NTOK / 8, 256, 0, stream>>>(Wh, a, s1, s2);

    // 4. N^2 leaky-relu row sums with s2 in LDS
    gsal_rowsums<<<NTOK / 256, 256, 0, stream>>>(s1, s2, rsum);

    // 5. softmax over the 8192 logits
    gsal_softmax<<<1, 1024, 0, stream>>>(rsum, att);

    // 6. deterministic attention-weighted column sums, then elu
    gsal_attn_partial<<<32, 256, 0, stream>>>(Wh, att, partial);
    gsal_finalize<<<1, 256, 0, stream>>>(partial, out);
}
